// OptimizedLSTM_42485816492100
// MI455X (gfx1250) — matrix operand order repaired and checked by reading, not compiled, not run on hardware
//
#include <hip/hip_runtime.h>
#include <hip/hip_bf16.h>

// ---------------------------------------------------------------------------
// 2-layer LSTM for MI455X (gfx1250, wave32, WMMA).
// gates = [x_t, h_{t-1}] @ W^T + b via v_wmma_f32_16x16x32_bf16 with bf16
// hi/lo operand splitting (fp32-class accuracy on the fast matrix pipe).
// All fp32->bf16 splitting is hoisted OUT of the GEMM loop: x is pre-split
// into A-fragments once; h is split by the previous step's epilogue. The
// inner loop is pure {fragment loads + WMMA}.
// ---------------------------------------------------------------------------

typedef __attribute__((ext_vector_type(16))) __bf16 v16bf;
typedef __attribute__((ext_vector_type(8)))  float  v8f;

#define B_   32
#define S_   512
#define IN_  512
#define H_   1024
#define H4_  4096
#define BH_  (B_ * H_)
#define NKTX0 (IN_ / 32)   // 16  x-part K tiles, layer 0
#define NKTH  (H_ / 32)    // 32  h-part K tiles (and layer-1 x-part)
#define FRAG  512          // elements per 16x32 fragment tile (32 lanes * 16)

// ---------------------------------------------------------------------------
// A-fragment layout (v_wmma 16x16x32, 16-bit A):
//   lane<16 : row = lane,    K = {0..7, 16..23}   (elements 0..7, 8..15)
//   lane>=16: row = lane-16, K = {8..15, 24..31}
// Fragment tile storage: [tile][lane][16 elems] contiguous bf16.
// B-fragment layout: lane<16: col N = lane, K = {0..7, 16..23} ; lane>=16: K = {8..15, 24..31}.
// ---------------------------------------------------------------------------

// Split fp32 W[4H, K] into bf16 hi/lo B-fragment tiles.
__global__ void prep_weights(const float* __restrict__ W,
                             __bf16* __restrict__ hi, __bf16* __restrict__ lo,
                             int nkt, int K) {
  const int lane = threadIdx.x & 31;
  const int wave = threadIdx.x >> 5;
  const int tile = blockIdx.x * 4 + wave;
  const int ct = tile / nkt;
  const int kt = tile % nkt;
  const int n = ct * 16 + (lane & 15);
  const int kbase = kt * 32 + ((lane & 16) ? 8 : 0);
  const float* src = W + (size_t)n * K + kbase;
  const size_t dst = (size_t)tile * FRAG + (size_t)lane * 16;
#pragma unroll
  for (int j = 0; j < 16; ++j) {
    float v = src[j < 8 ? j : j + 8];
    __bf16 h = (__bf16)v;
    hi[dst + j] = h;
    lo[dst + j] = (__bf16)(v - (float)h);
  }
}

// Split layer-0 input x [B, S, IN] into A-fragment tiles:
// layout [t][rb(2)][kt(NKTX0)][lane][16].
__global__ void prep_x(const float* __restrict__ x,
                       __bf16* __restrict__ hi, __bf16* __restrict__ lo) {
  const int lane = threadIdx.x & 31;
  const int wave = threadIdx.x >> 5;
  const int tile = blockIdx.x * 4 + wave;      // t*32 + rb*16 + kt
  const int kt = tile & 15;
  const int rb = (tile >> 4) & 1;
  const int t  = tile >> 5;
  const int rlo = lane & 15;
  const int khalf = (lane & 16) ? 8 : 0;
  const int row = rb * 16 + rlo;               // batch index
  const float* p = x + (size_t)row * S_ * IN_ + (size_t)t * IN_ + kt * 32 + khalf;
  float4 f0 = *reinterpret_cast<const float4*>(p);
  float4 f1 = *reinterpret_cast<const float4*>(p + 4);
  float4 f2 = *reinterpret_cast<const float4*>(p + 16);
  float4 f3 = *reinterpret_cast<const float4*>(p + 20);
  float v[16] = {f0.x, f0.y, f0.z, f0.w, f1.x, f1.y, f1.z, f1.w,
                 f2.x, f2.y, f2.z, f2.w, f3.x, f3.y, f3.z, f3.w};
  const size_t dst = (size_t)tile * FRAG + (size_t)lane * 16;
#pragma unroll
  for (int j = 0; j < 16; ++j) {
    __bf16 h = (__bf16)v[j];
    hi[dst + j] = h;
    lo[dst + j] = (__bf16)(v[j] - (float)h);
  }
}

__global__ void zero_u32(unsigned int* __restrict__ p, int n) {
  int i = blockIdx.x * blockDim.x + threadIdx.x;
  if (i < n) p[i] = 0u;
}

__device__ __forceinline__ v16bf ldfrag(const __bf16* __restrict__ base,
                                        int tile, int lane) {
  return *(reinterpret_cast<const v16bf*>(base + (size_t)tile * FRAG) + lane);
}

__device__ __forceinline__ float sigmoid_f(float x) {
  return 1.0f / (1.0f + __expf(-x));
}

// ---------------------------------------------------------------------------
// One LSTM timestep. Block = 128 thr = 4 waves; wave g computes gate g's
// [32 x 16] tile for this block's 16-column slice of H. Inner loop is pure
// fragment loads + WMMA (operands pre-split). Epilogue does the pointwise
// update and emits h_next both as fp32 (for finalize) and as pre-split
// bf16 A-fragments (for the next step / next layer).
// ---------------------------------------------------------------------------
__global__ void __launch_bounds__(128)
lstm_step(const __bf16* __restrict__ xf_hi, const __bf16* __restrict__ xf_lo,
          int nktx,
          const __bf16* __restrict__ hf_hi, const __bf16* __restrict__ hf_lo,
          __bf16* __restrict__ hfo_hi, __bf16* __restrict__ hfo_lo,
          __bf16* __restrict__ sqf_hi, __bf16* __restrict__ sqf_lo,
          float* __restrict__ cstate, float* __restrict__ hnext,
          const __bf16* __restrict__ wt_hi, const __bf16* __restrict__ wt_lo,
          const float* __restrict__ bias, int nktot) {
  __shared__ float gbuf[4][32][16];

  const int tid  = threadIdx.x;
  const int lane = tid & 31;
  const int gate = tid >> 5;                   // 0=i 1=f 2=g 3=o
  const int colBlock = blockIdx.x;             // 0..63
  const int ct = gate * (H_ / 16) + colBlock;

  const float bv = bias[gate * H_ + colBlock * 16 + (lane & 15)];
  v8f c0, c1;
#pragma unroll
  for (int i = 0; i < 8; ++i) { c0[i] = bv; c1[i] = bv; }

  const __bf16* wth = wt_hi + (size_t)ct * nktot * FRAG;
  const __bf16* wtl = wt_lo + (size_t)ct * nktot * FRAG;

  // x part
  for (int kt = 0; kt < nktx; ++kt) {
    v16bf a0h = ldfrag(xf_hi, kt, lane);
    v16bf a0l = ldfrag(xf_lo, kt, lane);
    v16bf a1h = ldfrag(xf_hi, nktx + kt, lane);
    v16bf a1l = ldfrag(xf_lo, nktx + kt, lane);
    v16bf bh  = ldfrag(wth, kt, lane);
    v16bf bl  = ldfrag(wtl, kt, lane);
    c0 = __builtin_amdgcn_wmma_f32_16x16x32_bf16(false, a0h, false, bh, (short)0, c0, false, false);
    c1 = __builtin_amdgcn_wmma_f32_16x16x32_bf16(false, a1h, false, bh, (short)0, c1, false, false);
    c0 = __builtin_amdgcn_wmma_f32_16x16x32_bf16(false, a0l, false, bh, (short)0, c0, false, false);
    c1 = __builtin_amdgcn_wmma_f32_16x16x32_bf16(false, a1l, false, bh, (short)0, c1, false, false);
    c0 = __builtin_amdgcn_wmma_f32_16x16x32_bf16(false, a0h, false, bl, (short)0, c0, false, false);
    c1 = __builtin_amdgcn_wmma_f32_16x16x32_bf16(false, a1h, false, bl, (short)0, c1, false, false);
  }
  // h part (fragments laid out [rb(2)][NKTH][lane][16])
  for (int kt = 0; kt < NKTH; ++kt) {
    v16bf a0h = ldfrag(hf_hi, kt, lane);
    v16bf a0l = ldfrag(hf_lo, kt, lane);
    v16bf a1h = ldfrag(hf_hi, NKTH + kt, lane);
    v16bf a1l = ldfrag(hf_lo, NKTH + kt, lane);
    v16bf bh  = ldfrag(wth, nktx + kt, lane);
    v16bf bl  = ldfrag(wtl, nktx + kt, lane);
    c0 = __builtin_amdgcn_wmma_f32_16x16x32_bf16(false, a0h, false, bh, (short)0, c0, false, false);
    c1 = __builtin_amdgcn_wmma_f32_16x16x32_bf16(false, a1h, false, bh, (short)0, c1, false, false);
    c0 = __builtin_amdgcn_wmma_f32_16x16x32_bf16(false, a0l, false, bh, (short)0, c0, false, false);
    c1 = __builtin_amdgcn_wmma_f32_16x16x32_bf16(false, a1l, false, bh, (short)0, c1, false, false);
    c0 = __builtin_amdgcn_wmma_f32_16x16x32_bf16(false, a0h, false, bl, (short)0, c0, false, false);
    c1 = __builtin_amdgcn_wmma_f32_16x16x32_bf16(false, a1h, false, bl, (short)0, c1, false, false);
  }

  // C/D layout: N = lane%16; VGPR v -> M=v (lanes<16) or M=v+8 (lanes>=16).
  {
    const int col  = lane & 15;
    const int moff = (lane & 16) ? 8 : 0;
#pragma unroll
    for (int v = 0; v < 8; ++v) gbuf[gate][moff + v][col] = c0[v];
#pragma unroll
    for (int v = 0; v < 8; ++v) gbuf[gate][16 + moff + v][col] = c1[v];
  }
  __syncthreads();

  // Pointwise: 512 elems / 128 threads.
#pragma unroll
  for (int rep = 0; rep < 4; ++rep) {
    const int lin = rep * 128 + tid;
    const int row = lin >> 4;                  // batch
    const int col = lin & 15;
    const float iv = sigmoid_f(gbuf[0][row][col]);
    const float fv = sigmoid_f(gbuf[1][row][col]);
    const float gv = tanhf(gbuf[2][row][col]);
    const float ov = sigmoid_f(gbuf[3][row][col]);
    const int hidx = colBlock * 16 + col;
    const size_t sidx = (size_t)row * H_ + hidx;
    const float cp = cstate[sidx];
    const float cn = fv * cp + iv * gv;
    const float hn = ov * tanhf(cn);
    cstate[sidx] = cn;
    hnext[sidx]  = hn;

    // Scatter h_next into A-fragment layout (pre-split for next consumers).
    const int kk  = hidx & 31;
    const int kt  = hidx >> 5;
    const int lhi = (kk >> 3) & 1;                     // lane half
    const int elem = (kk & 7) | ((kk & 16) >> 1);      // element 0..15
    const int rb  = row >> 4;
    const size_t foff =
        (((size_t)(rb * NKTH + kt)) * 32 + (lhi * 16 + (row & 15))) * 16 + elem;
    const __bf16 hh = (__bf16)hn;
    const __bf16 hl = (__bf16)(hn - (float)hh);
    hfo_hi[foff] = hh;
    hfo_lo[foff] = hl;
    if (sqf_hi) { sqf_hi[foff] = hh; sqf_lo[foff] = hl; }
  }
}

__global__ void finalize(const float* __restrict__ h0f, const float* __restrict__ h1f,
                         const float* __restrict__ c0s, const float* __restrict__ c1s,
                         float* __restrict__ out) {
  int i = blockIdx.x * blockDim.x + threadIdx.x;
  if (i < BH_) {
    float h1 = h1f[i];
    out[i] = h1;                 // out1[:, -1, :]
    out[BH_ + i] = h0f[i];       // h_n[0]
    out[2 * BH_ + i] = h1;       // h_n[1]
    out[3 * BH_ + i] = c0s[i];   // c_n[0]
    out[4 * BH_ + i] = c1s[i];   // c_n[1]
  }
}

extern "C" void kernel_launch(void* const* d_in, const int* in_sizes, int n_in,
                              void* d_out, int out_size, void* d_ws, size_t ws_size,
                              hipStream_t stream) {
  const float* x  = (const float*)d_in[0];
  const float* W0 = (const float*)d_in[1];
  const float* b0 = (const float*)d_in[2];
  const float* W1 = (const float*)d_in[3];
  const float* b1 = (const float*)d_in[4];
  float* out = (float*)d_out;

  const int K0 = IN_ + H_;      // 1536
  const int K1 = 2 * H_;        // 2048
  const int nktot0 = K0 / 32;   // 48
  const int nktot1 = K1 / 32;   // 64

  const size_t hfragElems = (size_t)2 * NKTH * FRAG;           // 32768 per buffer
  const size_t xfrag0Elems = (size_t)S_ * 2 * NKTX0 * FRAG;    // x layer0 fragments
  const size_t sqfragElems = (size_t)S_ * 2 * NKTH * FRAG;     // layer1 x fragments

  char* ws = (char*)d_ws;
  size_t o = 0;
  __bf16* wt0_hi = (__bf16*)(ws + o); o += (size_t)H4_ * K0 * 2;
  __bf16* wt0_lo = (__bf16*)(ws + o); o += (size_t)H4_ * K0 * 2;
  __bf16* wt1_hi = (__bf16*)(ws + o); o += (size_t)H4_ * K1 * 2;
  __bf16* wt1_lo = (__bf16*)(ws + o); o += (size_t)H4_ * K1 * 2;
  __bf16* xf_hi  = (__bf16*)(ws + o); o += xfrag0Elems * 2;
  __bf16* xf_lo  = (__bf16*)(ws + o); o += xfrag0Elems * 2;
  __bf16* sq_hi  = (__bf16*)(ws + o); o += sqfragElems * 2;
  __bf16* sq_lo  = (__bf16*)(ws + o); o += sqfragElems * 2;
  // Zeroed region: fp32 states then h-fragment ping-pong buffers.
  float* zbase = (float*)(ws + o);
  float* h0a = (float*)(ws + o); o += (size_t)BH_ * 4;
  float* h0b = (float*)(ws + o); o += (size_t)BH_ * 4;
  float* h1a = (float*)(ws + o); o += (size_t)BH_ * 4;
  float* h1b = (float*)(ws + o); o += (size_t)BH_ * 4;
  float* c0s = (float*)(ws + o); o += (size_t)BH_ * 4;
  float* c1s = (float*)(ws + o); o += (size_t)BH_ * 4;
  __bf16* hf0a_hi = (__bf16*)(ws + o); o += hfragElems * 2;
  __bf16* hf0a_lo = (__bf16*)(ws + o); o += hfragElems * 2;
  __bf16* hf0b_hi = (__bf16*)(ws + o); o += hfragElems * 2;
  __bf16* hf0b_lo = (__bf16*)(ws + o); o += hfragElems * 2;
  __bf16* hf1a_hi = (__bf16*)(ws + o); o += hfragElems * 2;
  __bf16* hf1a_lo = (__bf16*)(ws + o); o += hfragElems * 2;
  __bf16* hf1b_hi = (__bf16*)(ws + o); o += hfragElems * 2;
  __bf16* hf1b_lo = (__bf16*)(ws + o); o += hfragElems * 2;
  const int zeroWords = (int)(((char*)(ws + o) - (char*)zbase) / 4);

  // One-shot prep (deterministic every call).
  const int tiles0 = (H4_ / 16) * nktot0;
  const int tiles1 = (H4_ / 16) * nktot1;
  prep_weights<<<dim3(tiles0 / 4), dim3(128), 0, stream>>>(W0, wt0_hi, wt0_lo, nktot0, K0);
  prep_weights<<<dim3(tiles1 / 4), dim3(128), 0, stream>>>(W1, wt1_hi, wt1_lo, nktot1, K1);
  const int xtiles = S_ * 2 * NKTX0;   // 16384
  prep_x<<<dim3(xtiles / 4), dim3(128), 0, stream>>>(x, xf_hi, xf_lo);
  zero_u32<<<dim3((zeroWords + 255) / 256), dim3(256), 0, stream>>>((unsigned int*)zbase, zeroWords);

  // Layer 0
  for (int t = 0; t < S_; ++t) {
    const __bf16* hrh = (t & 1) ? hf0b_hi : hf0a_hi;
    const __bf16* hrl = (t & 1) ? hf0b_lo : hf0a_lo;
    __bf16* hwh = (t & 1) ? hf0a_hi : hf0b_hi;
    __bf16* hwl = (t & 1) ? hf0a_lo : hf0b_lo;
    float* hw = (t & 1) ? h0a : h0b;
    lstm_step<<<dim3(H_ / 16), dim3(128), 0, stream>>>(
        xf_hi + (size_t)t * 2 * NKTX0 * FRAG, xf_lo + (size_t)t * 2 * NKTX0 * FRAG, NKTX0,
        hrh, hrl, hwh, hwl,
        sq_hi + (size_t)t * 2 * NKTH * FRAG, sq_lo + (size_t)t * 2 * NKTH * FRAG,
        c0s, hw, wt0_hi, wt0_lo, b0, nktot0);
  }
  // Layer 1 (x part = layer-0 h fragments)
  for (int t = 0; t < S_; ++t) {
    const __bf16* hrh = (t & 1) ? hf1b_hi : hf1a_hi;
    const __bf16* hrl = (t & 1) ? hf1b_lo : hf1a_lo;
    __bf16* hwh = (t & 1) ? hf1a_hi : hf1b_hi;
    __bf16* hwl = (t & 1) ? hf1a_lo : hf1b_lo;
    float* hw = (t & 1) ? h1a : h1b;
    lstm_step<<<dim3(H_ / 16), dim3(128), 0, stream>>>(
        sq_hi + (size_t)t * 2 * NKTH * FRAG, sq_lo + (size_t)t * 2 * NKTH * FRAG, NKTH,
        hrh, hrl, hwh, hwl,
        (__bf16*)nullptr, (__bf16*)nullptr,
        c1s, hw, wt1_hi, wt1_lo, b1, nktot1);
  }
  // S even: final h is in the "a" fp32 buffers.
  finalize<<<dim3((BH_ + 255) / 256), dim3(256), 0, stream>>>(h0a, h1a, c0s, c1s, out);
}
